// AAttnFusion_39109972197392
// MI455X (gfx1250) — matrix operand order repaired and checked by reading, not compiled, not run on hardware
//
#include <hip/hip_runtime.h>
#include <hip/hip_bf16.h>

// ---------------------------------------------------------------------------
// MI455X (gfx1250, wave32) implementation of the AAttn fusion block.
// All GEMMs (conv1x1s + attention einsums) run on v_wmma_f32_16x16x32_f16.
// Attention V tiles are staged LDS-side by the Tensor Data Mover
// (tensor_load_to_lds + s_wait_tensorcnt).
//
// Workspace layout (bytes):
//   [ 0MB,  4MB)  x16    : f16 (B,128,4096)  silu(in-proj)
//   [ 4MB, 16MB)  qkv16  : f16 (B,384,4096)
//   [16MB, 24MB)  pp     : f32 (B,128,4096)  depthwise-7x7 PE
//   [24MB, 28MB)  y16    : f16 (B,128,4096)  attn_out + pp
//   [28MB, 36MB)  g16    : f16 (B,256,4096)  gelu(proj(r)) ++ gelu(proj(c))
// ---------------------------------------------------------------------------

typedef __attribute__((ext_vector_type(16))) _Float16 v16h;
typedef __attribute__((ext_vector_type(8)))  float    v8f;
typedef __attribute__((ext_vector_type(4)))  unsigned u32x4;
typedef __attribute__((ext_vector_type(8)))  int      i32x8;
typedef __attribute__((ext_vector_type(4)))  int      i32x4;

union Frag { unsigned u[8]; v16h h; };
static_assert(sizeof(Frag) == 32, "frag size");

#define ACT_NONE 0
#define ACT_SILU 1
#define ACT_GELU 2

__device__ __forceinline__ float apply_act(float x, int act) {
  if (act == ACT_SILU) return x / (1.0f + __expf(-x));
  if (act == ACT_GELU) return 0.5f * x * (1.0f + erff(x * 0.70710678118654752f));
  return x;
}

// ---------------------------------------------------------------------------
// Tensor Data Mover: 2D f16 tile (tile_w x tile_h, row stride in elements)
// from global memory into contiguous LDS at lds_off.  D# per ISA §8.3/8.4:
// group0 = {count=1 | lds_addr | global_addr | type=2}
// group1 = {mask=0, data_size=1(2B), dims, tile dims, dim0 stride}
// groups 2/3 + extra group zero (2D tensor, tile_dim2=0 -> unused).
// Toolchain here is the 6-arg clang-23 form:
//   (uint32x4, int32x8, int32x4, int32x4, int32x8, i32 cpol)
// ---------------------------------------------------------------------------
__device__ __forceinline__ void tdm_load_2d_f16(
    unsigned lds_off, unsigned long long gaddr,
    unsigned tile_w, unsigned tile_h, unsigned long long row_stride_elems)
{
  const unsigned td0 = 0x01000000u;  // generous tensor_dim0 (elements)
  const unsigned td1 = 0x01000000u;  // generous tensor_dim1
  u32x4 g0;
  g0.x = 0x1u;                                            // count=1, user D#
  g0.y = lds_off;                                         // lds_addr (bytes)
  g0.z = (unsigned)(gaddr & 0xFFFFFFFFull);               // global_addr[31:0]
  g0.w = (unsigned)((gaddr >> 32) & 0x01FFFFFFull) | (2u << 30);  // [56:32]|type=2
  i32x8 g1;
  g1[0] = (int)(1u << 16);                                // data_size=1 (2 bytes)
  g1[1] = (int)((td0 & 0xFFFFu) << 16);                   // tensor_dim0 lo16
  g1[2] = (int)(((td0 >> 16) & 0xFFFFu) | ((td1 & 0xFFFFu) << 16));
  g1[3] = (int)(((td1 >> 16) & 0xFFFFu) | (tile_w << 16));// tile_dim0
  g1[4] = (int)(tile_h & 0xFFFFu);                        // tile_dim1 (tile_dim2=0)
  g1[5] = (int)(row_stride_elems & 0xFFFFFFFFull);        // dim0_stride[31:0]
  g1[6] = (int)((row_stride_elems >> 32) & 0xFFFFull);    // dim0_stride[47:32]
  g1[7] = 0;
  i32x4 gz4 = {0, 0, 0, 0};
  i32x8 gz8 = {0, 0, 0, 0, 0, 0, 0, 0};
  __builtin_amdgcn_tensor_load_to_lds(g0, g1, gz4, gz4, gz8, 0); // TENSORcnt++
}

// A-fragment (16x32 f16): lane m = lane&15, half h = lane>>4.
// VGPR j -> K = 16h+2j (contiguous order, same as B).
// B-fragment (32x16 f16): lane col, VGPR j -> K = 16h+2j.

// ---------------------------------------------------------------------------
// Generic 64x64-tile GEMM:  out[b, coff+o, n] = act( sum_c W[o,c]*X[b,c,n] + bias[o] )
// X: (B, C, 4096) of TIN (float or _Float16).  W: (O, C) f32.  N fixed = 4096.
// ---------------------------------------------------------------------------
template <typename TIN>
__global__ __launch_bounds__(256) void gemm64x64_wmma(
    const TIN* __restrict__ X, const float* __restrict__ W,
    const float* __restrict__ bias,
    _Float16* __restrict__ out16, float* __restrict__ out32,
    int C, int Ctot, int coff, int act)
{
  const int tid  = threadIdx.x;
  const int lane = tid & 31;
  const int wave = tid >> 5;
  const int nB   = blockIdx.x * 64;
  const int oB   = blockIdx.y * 64;
  const int b    = blockIdx.z;

  __shared__ _Float16 wl[64 * 36];   // [o][k], stride 36 halfs (72B, dword aligned)
  __shared__ _Float16 xl[64 * 36];   // [n][k] (transposed X tile)

  v8f acc[2] = {};
  const size_t xbase = (size_t)b * C * 4096;
  const int m = lane & 15, h = lane >> 4;

  for (int k0 = 0; k0 < C; k0 += 32) {
    __syncthreads();
    {
      // stage W tile: 64 x 32, 8 f32->f16 per thread
      int o  = tid >> 2, kc = (tid & 3) * 8;
      const float* wp = W + (size_t)(oB + o) * C + k0 + kc;
      #pragma unroll
      for (int i = 0; i < 8; ++i) wl[o * 36 + kc + i] = (_Float16)wp[i];
      // stage X tile transposed: 32(k) x 64(n) -> xl[n][k]
      int k = tid >> 3, n0 = (tid & 7) * 8;
      const TIN* xp = X + xbase + (size_t)(k0 + k) * 4096 + nB + n0;
      #pragma unroll
      for (int i = 0; i < 8; ++i) xl[(n0 + i) * 36 + k] = (_Float16)xp[i];
      if (k0 + 32 < C) {  // hint next K tile toward L2 (global_prefetch_b8)
        __builtin_prefetch(wp + 32, 0, 0);
        __builtin_prefetch(xp + 32 * 4096, 0, 0);
      }
    }
    __syncthreads();

    #pragma unroll
    for (int ti = 0; ti < 2; ++ti) {
      int tile = wave * 2 + ti, ot = tile >> 2, nt = tile & 3;
      Frag af, bf;
      #pragma unroll
      for (int j = 0; j < 8; ++j) {
        int kk = 16 * h + 2 * j;
        af.u[j] = *(const unsigned*)((const char*)wl + (((ot * 16 + m) * 36 + kk) * 2));
        int kb = 16 * h + 2 * j;
        bf.u[j] = *(const unsigned*)((const char*)xl + (((nt * 16 + m) * 36 + kb) * 2));
      }
      acc[ti] = __builtin_amdgcn_wmma_f32_16x16x32_f16(
          false, af.h, false, bf.h, (short)0, acc[ti], false, false);
    }
  }

  #pragma unroll
  for (int ti = 0; ti < 2; ++ti) {
    int tile = wave * 2 + ti, ot = tile >> 2, nt = tile & 3;
    int ncol = nB + nt * 16 + m;
    #pragma unroll
    for (int r = 0; r < 8; ++r) {
      int orow = oB + ot * 16 + r + 8 * h;     // C/D layout: M = r + 8*half
      float v  = apply_act(acc[ti][r] + bias[orow], act);
      size_t oi = ((size_t)b * Ctot + coff + orow) * 4096 + ncol;
      if (out16) out16[oi] = (_Float16)v;
      if (out32) out32[oi] = v;
    }
  }
}

// ---------------------------------------------------------------------------
// Depthwise 7x7 conv (padding 3) on v4 = qkv channels [256,384), f32 out.
// ---------------------------------------------------------------------------
__global__ __launch_bounds__(256) void dwconv7x7(
    const _Float16* __restrict__ qkv, const float* __restrict__ w,
    const float* __restrict__ bias, float* __restrict__ pp)
{
  const int tid = threadIdx.x;
  const int bx = blockIdx.x, by = blockIdx.y;
  const int ch = blockIdx.z & 127, b = blockIdx.z >> 7;

  __shared__ _Float16 tile[22 * 24];
  __shared__ float    wl[49];

  const _Float16* src = qkv + ((size_t)b * 384 + 256 + ch) * 4096;
  for (int idx = tid; idx < 22 * 22; idx += 256) {
    int yy = idx / 22, xx = idx - yy * 22;
    int gy = by * 16 + yy - 3, gx = bx * 16 + xx - 3;
    _Float16 v = (_Float16)0.0f;
    if (gy >= 0 && gy < 64 && gx >= 0 && gx < 64) v = src[gy * 64 + gx];
    tile[yy * 24 + xx] = v;
  }
  if (tid < 49) wl[tid] = w[ch * 49 + tid];
  __syncthreads();

  int tx = tid & 15, ty = tid >> 4;
  float acc = bias[ch];
  #pragma unroll
  for (int ky = 0; ky < 7; ++ky)
    #pragma unroll
    for (int kx = 0; kx < 7; ++kx)
      acc += (float)tile[(ty + ky) * 24 + tx + kx] * wl[ky * 7 + kx];

  pp[((size_t)b * 128 + ch) * 4096 + (by * 16 + ty) * 64 + bx * 16 + tx] = acc;
}

// ---------------------------------------------------------------------------
// Area attention (flash style), one (ba, head, 64-query block) per workgroup.
// The reference's (B,C,N)->(Ba,C,Na) reshape collapses to flat rows:
//   (t>>2)*4096 + (t&3)*1024 == t*1024,  t = a*C2 + c2
// so q/k/v/out rows are contiguous 1024-half segments at row index t.
// V tiles (32 x 64 f16, row stride 1024) are DMA'd to LDS by the TDM.
// ---------------------------------------------------------------------------
__global__ __launch_bounds__(256) void area_attn_wmma(
    const _Float16* __restrict__ qkv, const float* __restrict__ pp,
    _Float16* __restrict__ y16)
{
  const int tid  = threadIdx.x;
  const int lane = tid & 31;
  const int wave = tid >> 5;
  const int qb   = blockIdx.x;        // query block 0..15
  const int hd_  = blockIdx.y;        // head 0..3
  const int baz  = blockIdx.z;        // ba 0..15
  const int b = baz >> 2, a = baz & 3;

  __shared__ _Float16 qT[64 * 36];    // [n][d]
  __shared__ _Float16 kT[64 * 36];    // [m][d]
  __shared__ _Float16 vL[32 * 64];    // [d][m], contiguous: filled by TDM
  __shared__ float    sS[64 * 66];    // raw scores [n][m]
  __shared__ _Float16 pL[64 * 72];    // softmax probs [n][m]
  __shared__ float    corrL[64];
  __shared__ float    lF[64];

  // flat row bases (in halfs); row d is at base + d*1024
  const size_t qbase = (size_t)b * 1572864 + (size_t)(a * 256 + hd_ * 32) * 1024;
  const size_t kbase = qbase + (size_t)128 * 1024;
  const size_t vbase = (size_t)b * 1572864 + (size_t)256 * 4096 +
                       (size_t)(a * 128 + hd_ * 32) * 1024;
  const unsigned long long vgaddr =
      (unsigned long long)(uintptr_t)(qkv + vbase);      // bytes, tile row 0
  const unsigned vlds = (unsigned)(uintptr_t)(void*)vL;  // LDS byte offset

  { // stage Q^T once: d = tid>>3 (0..31), 8 contiguous n per thread
    int d = tid >> 3, n0 = (tid & 7) * 8;
    const _Float16* src = qkv + qbase + (size_t)d * 1024 + qb * 64 + n0;
    #pragma unroll
    for (int i = 0; i < 8; ++i) qT[(n0 + i) * 36 + d] = src[i];
  }

  const int m = lane & 15, h = lane >> 4;
  const int dt = wave >> 2, nt = wave & 3;  // this wave's O tile (2 x 4 tiles)
  float mrun = -3.0e38f, lrun = 0.0f;       // live in threads tid<64
  v8f oacc = {};

  for (int jb = 0; jb < 16; ++jb) {
    __syncthreads();                         // prior-iter LDS reads complete
    if (wave == 0) {                         // TDM: V tile -> vL (32x64, stride 1024)
      tdm_load_2d_f16(vlds, vgaddr + (unsigned long long)jb * 128, 64u, 32u, 1024ull);
    }
    { // stage K^T (transpose scatter) for this key block
      int d = tid >> 3, m0 = (tid & 7) * 8;
      const _Float16* sk = qkv + kbase + (size_t)d * 1024 + jb * 64 + m0;
      #pragma unroll
      for (int i = 0; i < 8; ++i) kT[(m0 + i) * 36 + d] = sk[i];
    }
    if (wave == 0) __builtin_amdgcn_s_wait_tensorcnt(0); // V tile landed
    __syncthreads();                         // publish kT + vL to all waves

    // S = Q^T K  (K-dim = hd = 32 -> single wmma per 16x16 tile; 2 tiles/wave)
    #pragma unroll
    for (int ti = 0; ti < 2; ++ti) {
      int tile = wave * 2 + ti, snt = tile >> 2, smt = tile & 3;
      Frag af, bf;
      #pragma unroll
      for (int j = 0; j < 8; ++j) {
        int kk = 16 * h + 2 * j;
        af.u[j] = *(const unsigned*)((const char*)qT + (((snt * 16 + m) * 36 + kk) * 2));
        int kb = 16 * h + 2 * j;
        bf.u[j] = *(const unsigned*)((const char*)kT + (((smt * 16 + m) * 36 + kb) * 2));
      }
      v8f sacc = {};
      sacc = __builtin_amdgcn_wmma_f32_16x16x32_f16(
          false, af.h, false, bf.h, (short)0, sacc, false, false);
      #pragma unroll
      for (int r = 0; r < 8; ++r)
        sS[(snt * 16 + r + 8 * h) * 66 + smt * 16 + m] = sacc[r] * 0.17677669529663687f;
    }
    __syncthreads();

    // Online softmax: one thread per query row
    if (tid < 64) {
      const float* srow = &sS[tid * 66];
      float mx = -3.0e38f;
      #pragma unroll 8
      for (int mm = 0; mm < 64; ++mm) mx = fmaxf(mx, srow[mm]);
      float mnew = fmaxf(mrun, mx);
      float corr = __expf(mrun - mnew);   // 0 on first block
      float sum = 0.0f;
      _Float16* prow = &pL[tid * 72];
      #pragma unroll 8
      for (int mm = 0; mm < 64; ++mm) {
        float e = __expf(srow[mm] - mnew);
        prow[mm] = (_Float16)e;
        sum += e;
      }
      lrun = lrun * corr + sum;
      mrun = mnew;
      corrL[tid] = corr;
    }
    __syncthreads();

    // Rescale running accumulator, then O += V * P^T  (K = 64 -> 2 wmma)
    {
      float cf = corrL[nt * 16 + m];
      #pragma unroll
      for (int r = 0; r < 8; ++r) oacc[r] *= cf;
      #pragma unroll
      for (int ko = 0; ko < 64; ko += 32) {
        Frag af, bf;
        #pragma unroll
        for (int j = 0; j < 8; ++j) {
          int kk = 16 * h + 2 * j;
          af.u[j] = *(const unsigned*)((const char*)vL + (((dt * 16 + m) * 64 + ko + kk) * 2));
          int kb = 16 * h + 2 * j;
          bf.u[j] = *(const unsigned*)((const char*)pL + (((nt * 16 + m) * 72 + ko + kb) * 2));
        }
        oacc = __builtin_amdgcn_wmma_f32_16x16x32_f16(
            false, af.h, false, bf.h, (short)0, oacc, false, false);
      }
    }
  }

  if (tid < 64) lF[tid] = lrun;
  __syncthreads();

  { // finalize: /l, flat-row store, fuse +pp, write f16
    float inv = 1.0f / lF[nt * 16 + m];
    int na = qb * 64 + nt * 16 + m;
    #pragma unroll
    for (int r = 0; r < 8; ++r) {
      int d = dt * 16 + r + 8 * h;
      size_t idx = (size_t)b * 524288 +
                   (size_t)(a * 128 + hd_ * 32 + d) * 1024 + na;
      y16[idx] = (_Float16)(oacc[r] * inv + pp[idx]);
    }
  }
}

// ---------------------------------------------------------------------------
extern "C" void kernel_launch(void* const* d_in, const int* in_sizes, int n_in,
                              void* d_out, int out_size, void* d_ws, size_t ws_size,
                              hipStream_t stream) {
  (void)in_sizes; (void)n_in; (void)out_size; (void)ws_size;

  const float* rgb    = (const float*)d_in[0];
  const float* chm    = (const float*)d_in[1];
  const float* w1[2]  = {(const float*)d_in[2],  (const float*)d_in[4]};
  const float* b1[2]  = {(const float*)d_in[3],  (const float*)d_in[5]};
  const float* fuse_w = (const float*)d_in[6];
  const float* fuse_b = (const float*)d_in[7];
  const float* qkv_w[2]  = {(const float*)d_in[8],  (const float*)d_in[14]};
  const float* qkv_b[2]  = {(const float*)d_in[9],  (const float*)d_in[15]};
  const float* pe_w[2]   = {(const float*)d_in[10], (const float*)d_in[16]};
  const float* pe_b[2]   = {(const float*)d_in[11], (const float*)d_in[17]};
  const float* proj_w[2] = {(const float*)d_in[12], (const float*)d_in[18]};
  const float* proj_b[2] = {(const float*)d_in[13], (const float*)d_in[19]};
  const float* x_in[2]   = {rgb, chm};

  char* ws = (char*)d_ws;
  _Float16* x16   = (_Float16*)(ws);
  _Float16* qkv16 = (_Float16*)(ws + ( 4u << 20));
  float*    pp    = (float*)   (ws + (16u << 20));
  _Float16* y16   = (_Float16*)(ws + (24u << 20));
  _Float16* g16   = (_Float16*)(ws + (28u << 20));

  dim3 blk(256);
  for (int mod = 0; mod < 2; ++mod) {
    // x = silu(conv1x1(in, w1))            (B,256,N) -> f16 (B,128,N)
    gemm64x64_wmma<float><<<dim3(64, 2, 4), blk, 0, stream>>>(
        x_in[mod], w1[mod], b1[mod], x16, nullptr, 256, 128, 0, ACT_SILU);
    // qkv = conv1x1(x, qkv_w)              f16 (B,384,N)
    gemm64x64_wmma<_Float16><<<dim3(64, 6, 4), blk, 0, stream>>>(
        x16, qkv_w[mod], qkv_b[mod], qkv16, nullptr, 128, 384, 0, ACT_NONE);
    // pp = dwconv7(v4)                     f32 (B,128,N)
    dwconv7x7<<<dim3(4, 4, 512), blk, 0, stream>>>(qkv16, pe_w[mod], pe_b[mod], pp);
    // y = attn(q,k,v) + pp                 f16 (B,128,N)
    area_attn_wmma<<<dim3(16, 4, 16), blk, 0, stream>>>(qkv16, pp, y16);
    // g[:, mod*128:+128] = gelu(conv1x1(y, proj_w))
    gemm64x64_wmma<_Float16><<<dim3(64, 2, 4), blk, 0, stream>>>(
        y16, proj_w[mod], proj_b[mod], g16, nullptr, 128, 256, mod * 128, ACT_GELU);
  }
  // out = silu(conv1x1(g, fuse_w))         f32 (B,256,N) -> d_out
  gemm64x64_wmma<_Float16><<<dim3(64, 4, 4), blk, 0, stream>>>(
      g16, fuse_w, fuse_b, nullptr, (float*)d_out, 256, 256, 0, ACT_SILU);
}
